// GCRN_29265907155019
// MI455X (gfx1250) — matrix operand order repaired and checked by reading, not compiled, not run on hardware
//
#include <hip/hip_runtime.h>

// ---------------------------------------------------------------------------
// GCRN for MI455X (gfx1250): GCN scatter (L2 atomics) + GRU via v_wmma f16
// ---------------------------------------------------------------------------

typedef __attribute__((ext_vector_type(16))) _Float16 v16h;
typedef __attribute__((ext_vector_type(2)))  _Float16 h2v;
typedef __attribute__((ext_vector_type(8)))  _Float16 v8h;
typedef __attribute__((ext_vector_type(8)))  float    v8f;

#define N_NODES  20000
#define N_EDGES  640000
#define HISTORY  24
#define HIDDEN   64
#define GATES    192            // 3 * 64
#define TILES_M  (N_NODES / 16) // 1250
#define WPB      8              // waves per block

#define WMMA_F16(A, B, C) \
  __builtin_amdgcn_wmma_f32_16x16x32_f16(false, (A), false, (B), (short)0, (C), false, false)

// fast sigmoid / tanh: single v_exp_f32 + single v_rcp_f32 (no IEEE div fixup)
static __device__ __forceinline__ float sigm(float x) {
  return __builtin_amdgcn_rcpf(1.f + __expf(-x));
}
static __device__ __forceinline__ float tanh_fast(float x) {
  float e2 = __expf(2.f * x);
  return 1.f - 2.f * __builtin_amdgcn_rcpf(e2 + 1.f); // safe at +-inf
}

// ---------------------------------------------------------------------------
__global__ void zero_f32_kernel(float* p, int n) {
  int i = blockIdx.x * blockDim.x + threadIdx.x;
  if (i < n) p[i] = 0.f;
}

__global__ void deg_kernel(const long long* __restrict__ ei, float* __restrict__ deg) {
  int e = blockIdx.x * blockDim.x + threadIdx.x;
  if (e < N_EDGES) atomicAdd(&deg[(int)ei[N_EDGES + e]], 1.0f);
}

__global__ void dinv_kernel(float* __restrict__ deg) {
  int i = blockIdx.x * blockDim.x + threadIdx.x;
  if (i < N_NODES) {
    float d = deg[i] + 1.0f; // + self loop, always > 0
    deg[i] = rsqrtf(d);      // in-place: deg -> dinv
  }
}

// agg[n][t] = x[n][t] * dinv[n]^2   (self-loop term, also initializes buffer)
__global__ void agg_init_kernel(const float* __restrict__ x, const float* __restrict__ dinv,
                                float* __restrict__ agg) {
  int i = blockIdx.x * blockDim.x + threadIdx.x;
  if (i < N_NODES * HISTORY) {
    int node = i / HISTORY;
    float w = dinv[node] * dinv[node];
    agg[i] = x[i] * w;
  }
}

__global__ void agg_edges_kernel(const long long* __restrict__ ei, const float* __restrict__ x,
                                 const float* __restrict__ dinv, float* __restrict__ agg) {
  int e = blockIdx.x * blockDim.x + threadIdx.x;
  if (e >= N_EDGES) return;
  int s = (int)ei[e];
  int d = (int)ei[N_EDGES + e];
  float w = dinv[s] * dinv[d];
  const float* xs = x + (long)s * HISTORY;
  float* ad = agg + (long)d * HISTORY;
#pragma unroll
  for (int t = 0; t < HISTORY; ++t) atomicAdd(&ad[t], xs[t] * w);
}

// ---------------------------------------------------------------------------
// GRU kernel: 1 wave per 16-node tile, WMMA f32_16x16x32_f16.
// ---------------------------------------------------------------------------
__launch_bounds__(256)
__global__ void gru_kernel(const float* __restrict__ agg,
                           const float* __restrict__ W_in, const float* __restrict__ b_in,
                           const float* __restrict__ W_ih, const float* __restrict__ W_hh,
                           const float* __restrict__ b_ih, const float* __restrict__ b_hh,
                           const float* __restrict__ W_out, float* __restrict__ y) {
  // W in LDS as f16, [n (gate unit)][k (feature)] layout
  __shared__ _Float16 sWih[HIDDEN * GATES];            // 24 KB
  __shared__ _Float16 sWhh[HIDDEN * GATES];            // 24 KB
  __shared__ _Float16 sH[WPB][16 * HIDDEN];            // 16 KB, per-wave h staging

  const int tid = threadIdx.x;
  for (int i = tid; i < HIDDEN * GATES; i += 256) {
    int n = i / HIDDEN;      // gate-unit row of W (192)
    int k = i % HIDDEN;      // input feature col (64)
    sWih[n * HIDDEN + k] = (_Float16)W_ih[i];
    sWhh[n * HIDDEN + k] = (_Float16)W_hh[i];
  }
  __syncthreads();

  const int wave = tid >> 5;
  const int lane = tid & 31;
  const int tile = blockIdx.x * WPB + wave;
  if (tile >= TILES_M) return;
  const int node0 = tile * 16;
  const int mA = lane & 15;                  // A-layout row, also C/D column
  const int laneoff = (lane >= 16) ? 8 : 0;  // A-layout K offset for hi half-wave
  const int kb[8] = {0, 2, 4, 6, 16, 18, 20, 22};

  // per-lane W_in / b_in, packed as f16 pairs in A-fragment K order
  // hw[kc*8+v] = {W_in[K0], W_in[K0+1]},  K0 = kb[v] + laneoff + 32*kc
  h2v hw[16], hb[16];
#pragma unroll
  for (int kc = 0; kc < 2; ++kc)
#pragma unroll
    for (int v = 0; v < 8; ++v) {
      int K = kb[v] + laneoff + 32 * kc;
      int i = kc * 8 + v;
      hw[i][0] = (_Float16)W_in[K];     hw[i][1] = (_Float16)W_in[K + 1];
      hb[i][0] = (_Float16)b_in[K];     hb[i][1] = (_Float16)b_in[K + 1];
    }

  // per-lane gate biases (column = 16*j + mA of the 192-wide gate vector)
  float br4[4], bz4[4], bi4[4], bh4[4];
#pragma unroll
  for (int j = 0; j < 4; ++j) {
    int col = 16 * j + mA;
    br4[j] = b_ih[col] + b_hh[col];
    bz4[j] = b_ih[64 + col] + b_hh[64 + col];
    bi4[j] = b_ih[128 + col];
    bh4[j] = b_hh[128 + col];
  }

  _Float16* hW = sH[wave];
  const float* aggp = agg + (long)(node0 + mA) * HISTORY;

  // h state kept in registers in C/D tile layout: ht[j] covers hid cols 16j..16j+15
  v8f ht[4];
#pragma unroll
  for (int j = 0; j < 4; ++j)
#pragma unroll
    for (int v = 0; v < 8; ++v) ht[j][v] = 0.f;

  for (int t = 0; t < HISTORY; ++t) {
    // ---- stage h_{t-1} to LDS in [node][hid] order (f16) ----
    {
      int rbase = (lane < 16) ? 0 : 8;
#pragma unroll
      for (int j = 0; j < 4; ++j)
#pragma unroll
        for (int v = 0; v < 8; ++v)
          hW[(rbase + v) * HIDDEN + 16 * j + mA] = (_Float16)ht[j][v];
    }
    asm volatile("s_wait_dscnt 0" ::: "memory"); // cross-lane LDS store->load

    // ---- A fragments: x_t = relu(agg*W_in + b_in), packed f16 math ----
    float a = aggp[t];
    h2v aa; aa[0] = (_Float16)a; aa[1] = aa[0];
    h2v hzero; hzero[0] = (_Float16)0.f; hzero[1] = hzero[0];
    v16h ax0, ax1;
#pragma unroll
    for (int v = 0; v < 8; ++v) {
      h2v r0 = __builtin_elementwise_max((h2v)(aa * hw[v] + hb[v]),     hzero);
      h2v r1 = __builtin_elementwise_max((h2v)(aa * hw[8 + v] + hb[8 + v]), hzero);
      ax0[2 * v] = r0[0]; ax0[2 * v + 1] = r0[1];
      ax1[2 * v] = r1[0]; ax1[2 * v + 1] = r1[1];
    }
    // ---- A fragments of h from LDS ----
    const h2v* ph = (const h2v*)(hW + mA * HIDDEN);
    v16h ah0, ah1;
#pragma unroll
    for (int v = 0; v < 8; ++v) {
      int K0 = kb[v] + laneoff; // even
      h2v p0 = ph[K0 >> 1];
      h2v p1 = ph[(K0 + 32) >> 1];
      ah0[2 * v] = p0[0]; ah0[2 * v + 1] = p0[1];
      ah1[2 * v] = p1[0]; ah1[2 * v + 1] = p1[1];
    }

    // ---- gates, one 16-wide hid tile at a time ----
#pragma unroll
    for (int j = 0; j < 4; ++j) {
      // bias folded into accumulator init (C/D layout: one column per lane)
      v8f racc, zacc, iacc, hacc;
#pragma unroll
      for (int v = 0; v < 8; ++v) {
        racc[v] = br4[j]; zacc[v] = bz4[j]; iacc[v] = bi4[j]; hacc[v] = bh4[j];
      }

      // B fragment: lane = N column, halves = K in A-fragment (table) order
      #define LDB(Wp, kc, nt) (__builtin_shufflevector(*(const v8h*)((Wp) + ((nt) * 16 + mA) * HIDDEN + (kc) * 32 + laneoff), *(const v8h*)((Wp) + ((nt) * 16 + mA) * HIDDEN + (kc) * 32 + 16 + laneoff), 0, 1, 2, 3, 4, 5, 6, 7, 8, 9, 10, 11, 12, 13, 14, 15))
      // r gate: gi + gh accumulated in ONE chain (4 WMMAs, D->C pipelined)
      racc = WMMA_F16(ax0, LDB(sWih, 0, j),     racc);
      racc = WMMA_F16(ax1, LDB(sWih, 1, j),     racc);
      racc = WMMA_F16(ah0, LDB(sWhh, 0, j),     racc);
      racc = WMMA_F16(ah1, LDB(sWhh, 1, j),     racc);
      // z gate
      zacc = WMMA_F16(ax0, LDB(sWih, 0, 4 + j), zacc);
      zacc = WMMA_F16(ax1, LDB(sWih, 1, 4 + j), zacc);
      zacc = WMMA_F16(ah0, LDB(sWhh, 0, 4 + j), zacc);
      zacc = WMMA_F16(ah1, LDB(sWhh, 1, 4 + j), zacc);
      // n gate: ig and hg must stay separate (g = tanh(ig + r*hg))
      iacc = WMMA_F16(ax0, LDB(sWih, 0, 8 + j), iacc);
      iacc = WMMA_F16(ax1, LDB(sWih, 1, 8 + j), iacc);
      hacc = WMMA_F16(ah0, LDB(sWhh, 0, 8 + j), hacc);
      hacc = WMMA_F16(ah1, LDB(sWhh, 1, 8 + j), hacc);
      #undef LDB

#pragma unroll
      for (int v = 0; v < 8; ++v) {
        float r = sigm(racc[v]);
        float z = sigm(zacc[v]);
        float g = tanh_fast(fmaf(r, hacc[v], iacc[v]));
        ht[j][v] = fmaf(z, ht[j][v] - g, g); // (1-z)*g + z*h
      }
    }
  }

  // ---- stage final h, then y = h_last @ W_out ----
  {
    int rbase = (lane < 16) ? 0 : 8;
#pragma unroll
    for (int j = 0; j < 4; ++j)
#pragma unroll
      for (int v = 0; v < 8; ++v)
        hW[(rbase + v) * HIDDEN + 16 * j + mA] = (_Float16)ht[j][v];
  }
  asm volatile("s_wait_dscnt 0" ::: "memory");

  if (lane < 16) {
    const _Float16* hp = hW + lane * HIDDEN;
    float s = 0.f;
#pragma unroll
    for (int c = 0; c < HIDDEN; ++c) s += (float)hp[c] * W_out[c];
    y[node0 + lane] = s;
  }
}

// ---------------------------------------------------------------------------
__global__ void out_init_kernel(const float* __restrict__ y, const float* __restrict__ dinv,
                                const float* __restrict__ b_out, float* __restrict__ out) {
  int i = blockIdx.x * blockDim.x + threadIdx.x;
  if (i < N_NODES) out[i] = b_out[0] + y[i] * dinv[i] * dinv[i];
}

__global__ void out_edges_kernel(const long long* __restrict__ ei, const float* __restrict__ y,
                                 const float* __restrict__ dinv, float* __restrict__ out) {
  int e = blockIdx.x * blockDim.x + threadIdx.x;
  if (e >= N_EDGES) return;
  int s = (int)ei[e];
  int d = (int)ei[N_EDGES + e];
  atomicAdd(&out[d], y[s] * dinv[s] * dinv[d]);
}

// ---------------------------------------------------------------------------
extern "C" void kernel_launch(void* const* d_in, const int* in_sizes, int n_in,
                              void* d_out, int out_size, void* d_ws, size_t ws_size,
                              hipStream_t stream) {
  const float*      x_seq = (const float*)d_in[0];
  const long long*  ei    = (const long long*)d_in[1];
  const float*      W_in  = (const float*)d_in[2];
  const float*      b_in  = (const float*)d_in[3];
  const float*      W_ih  = (const float*)d_in[4];
  const float*      W_hh  = (const float*)d_in[5];
  const float*      b_ih  = (const float*)d_in[6];
  const float*      b_hh  = (const float*)d_in[7];
  const float*      W_out = (const float*)d_in[8];
  const float*      b_out = (const float*)d_in[9];
  float*            out   = (float*)d_out;

  float* dinv = (float*)d_ws;                       // N  (degree, then dinv in place)
  float* agg  = dinv + N_NODES;                     // N * 24
  float* yv   = agg + (size_t)N_NODES * HISTORY;    // N

  const int B = 256;
  zero_f32_kernel<<<(N_NODES + B - 1) / B, B, 0, stream>>>(dinv, N_NODES);
  deg_kernel<<<(N_EDGES + B - 1) / B, B, 0, stream>>>(ei, dinv);
  dinv_kernel<<<(N_NODES + B - 1) / B, B, 0, stream>>>(dinv);
  agg_init_kernel<<<(N_NODES * HISTORY + B - 1) / B, B, 0, stream>>>(x_seq, dinv, agg);
  agg_edges_kernel<<<(N_EDGES + B - 1) / B, B, 0, stream>>>(ei, x_seq, dinv, agg);
  gru_kernel<<<(TILES_M + WPB - 1) / WPB, 256, 0, stream>>>(agg, W_in, b_in, W_ih, W_hh,
                                                            b_ih, b_hh, W_out, yv);
  out_init_kernel<<<(N_NODES + B - 1) / B, B, 0, stream>>>(yv, dinv, b_out, out);
  out_edges_kernel<<<(N_EDGES + B - 1) / B, B, 0, stream>>>(ei, yv, dinv, out);
}
